// RSSM_77103252898295
// MI455X (gfx1250) — matrix operand order repaired and checked by reading, not compiled, not run on hardware
//
#include <hip/hip_runtime.h>
#include <math.h>

// ---------------------------------------------------------------------------
// Problem dimensions (fixed by the reference)
// ---------------------------------------------------------------------------
#define B_SZ 512
#define T_SZ 256
#define DXO  1024   // emitter output dim
#define DZD  256    // latent dim
#define DHD  512    // hidden dim
#define DDZ  256    // transition MLP hidden
#define DDX  256    // emitter MLP hidden

// Padded LDS row strides (bf16 elements): +16 elems (=32B) breaks the
// 256B (64 bank x 4B) conflict period while keeping 16B row alignment.
#define LDH (DHD + 16)   // 528
#define LDZ (DZD + 16)   // 272

typedef __attribute__((ext_vector_type(16))) __bf16 v16bf;
typedef __attribute__((ext_vector_type(8)))  __bf16 v8bf;
typedef __attribute__((ext_vector_type(8)))  float  v8f;

__device__ __forceinline__ float sigmoidf_(float x) { return 1.0f / (1.0f + __expf(-x)); }
__device__ __forceinline__ float softplusf_(float x) { return (x > 20.0f) ? x : log1pf(__expf(x)); }

// Opaque compile-time memory barrier: stops LICM from hoisting the (loop-
// invariant) weight streams out of the t-loop and spilling them to scratch.
__device__ __forceinline__ void antihoist() { asm volatile("" ::: "memory"); }

// A-fragment (16x32 bf16, M x K).  Lane L holds row M=L&15; lanes 0-15 hold
// K chunks {k0..k0+7, k0+16..k0+23}, lanes 16-31 the other halves.
__device__ __forceinline__ v16bf fragA(const __bf16* base, int m, int ld, int k0, int half) {
  const __bf16* p = base + m * ld + k0 + half * 8;
  v8bf lo = *(const v8bf*)(p);
  v8bf hi = *(const v8bf*)(p + 16);
  return __builtin_shufflevector(lo, hi, 0, 1, 2, 3, 4, 5, 6, 7, 8, 9, 10, 11, 12, 13, 14, 15);
}

// B-fragment (32x16 bf16, K x N) for X @ W^T with row-major W[out,in]:
// column n of B == row (col0+n) of W -> two 16B reads per lane, K chunks
// {k0+8g..k0+8g+7, k0+16+8g..k0+16+8g+7} (same order as fragA).
__device__ __forceinline__ v16bf fragB(const __bf16* w, int row, int K, int k0, int g) {
  const __bf16* p = w + (size_t)row * K + k0 + g * 8;
  v8bf lo = *(const v8bf*)(p);
  v8bf hi = *(const v8bf*)(p + 16);
  return __builtin_shufflevector(lo, hi, 0, 1, 2, 3, 4, 5, 6, 7, 8, 9, 10, 11, 12, 13, 14, 15);
}

__device__ __forceinline__ v8f wmma_bf(v16bf a, v16bf b, v8f c) {
  return __builtin_amdgcn_wmma_f32_16x16x32_bf16(false, a, false, b, (short)0, c, false, false);
}

// ---------------------------------------------------------------------------
// Prep: fp32 -> bf16 weight conversion into workspace (L2-resident).
// ---------------------------------------------------------------------------
__global__ void cvt_bf16_kernel(const float* __restrict__ s, __bf16* __restrict__ d, int n) {
  int i = blockIdx.x * blockDim.x + threadIdx.x;
  if (i < n) d[i] = (__bf16)s[i];
}

// ---------------------------------------------------------------------------
// Kernel 1: serial scan (phases 1-3).  One WG per 16-row batch tile loops
// over all T steps; h double-buffered in LDS; h_t streamed to global (bf16).
// Each A fragment is loaded ONCE per k-step and shared by all of the wave's
// accumulators (loop-interchanged vs the naive per-tile form).
// C-layout coords: m = i + 8*(lane>>4), n = lane&15.
// ---------------------------------------------------------------------------
__global__ __launch_bounds__(256, 1) void dmm_scan_kernel(
    const float* __restrict__ eps,                                       // [B,T,DZ]
    const __bf16* __restrict__ wz1,   const float* __restrict__ bz1,     // [DDZ,DH],[DDZ]
    const __bf16* __restrict__ wloc,  const float* __restrict__ bloc,    // [DZ,DDZ],[DZ]
    const __bf16* __restrict__ wscl,  const float* __restrict__ bscl,    // [DZ,DDZ],[DZ]
    const __bf16* __restrict__ wih,   const __bf16* __restrict__ whh,    // [3DH,DZ],[3DH,DH]
    const float* __restrict__ bih,    const float* __restrict__ bhh,     // [3DH],[3DH]
    const float* __restrict__ h0,                                        // [DH]
    __bf16* __restrict__ hseq)                                           // [B,T,DH] out
{
  __shared__ __bf16 sh_h[2][16 * LDH];     // double-buffered hidden state
  __shared__ __bf16 sh_hid[16 * LDZ];      // transition MLP hidden
  __shared__ __bf16 sh_z[16 * LDZ];        // latent sample

  const int tid  = threadIdx.x;
  const int lane = tid & 31;
  const int wave = tid >> 5;
  const int half = lane >> 4;
  const int nl   = lane & 15;
  const int b0   = blockIdx.x * 16;

  for (int k = tid; k < DHD; k += 256) {
    __bf16 v = (__bf16)h0[k];
    #pragma unroll
    for (int m = 0; m < 16; ++m) sh_h[0][m * LDH + k] = v;
  }
  __syncthreads();

  int cur = 0;
  for (int t = 0; t < T_SZ; ++t) {
    antihoist();
    const __bf16* hb = sh_h[cur];

    // -------- phase 1: hid = relu(h @ dz_W1^T + b1)   [16 x DDZ], K=DH ----
    {
      v8f acc[2] = {};
      #pragma unroll
      for (int k0 = 0; k0 < DHD; k0 += 32) {
        v16bf a = fragA(hb, nl, LDH, k0, half);
        #pragma unroll
        for (int jt = 0; jt < 2; ++jt)
          acc[jt] = wmma_bf(a, fragB(wz1, (wave * 2 + jt) * 16 + nl, DHD, k0, half), acc[jt]);
      }
      #pragma unroll
      for (int jt = 0; jt < 2; ++jt) {
        int col0 = (wave * 2 + jt) * 16;
        float bias = bz1[col0 + nl];
        #pragma unroll
        for (int i = 0; i < 8; ++i) {
          int m = i + 8 * half;
          float v = acc[jt][i] + bias;
          sh_hid[m * LDZ + col0 + nl] = (__bf16)(v > 0.0f ? v : 0.0f);
        }
      }
    }
    __syncthreads();

    // -------- phase 2: z = loc + softplus(scale_pre) * eps   [16 x DZ] ----
    {
      v8f aL[2] = {}, aS[2] = {};
      #pragma unroll
      for (int k0 = 0; k0 < DDZ; k0 += 32) {
        v16bf a = fragA(sh_hid, nl, LDZ, k0, half);
        #pragma unroll
        for (int jt = 0; jt < 2; ++jt) {
          int col = (wave * 2 + jt) * 16 + nl;
          aL[jt] = wmma_bf(a, fragB(wloc, col, DDZ, k0, half), aL[jt]);
          aS[jt] = wmma_bf(a, fragB(wscl, col, DDZ, k0, half), aS[jt]);
        }
      }
      #pragma unroll
      for (int jt = 0; jt < 2; ++jt) {
        int col0 = (wave * 2 + jt) * 16;
        float bl = bloc[col0 + nl], bs = bscl[col0 + nl];
        #pragma unroll
        for (int i = 0; i < 8; ++i) {
          int m = i + 8 * half;
          float e = eps[((size_t)(b0 + m) * T_SZ + t) * DZD + col0 + nl];
          float z = (aL[jt][i] + bl) + softplusf_(aS[jt][i] + bs) * e;
          sh_z[m * LDZ + col0 + nl] = (__bf16)z;
        }
      }
    }
    __syncthreads();

    // -------- phase 3: GRU.  Wave owns 4 column tiles; process in pairs so
    // one A-load feeds 6 WMMAs while staying under the VGPR ceiling. --------
    __bf16* hn = sh_h[cur ^ 1];
    for (int p = 0; p < 2; ++p) {
      v8f aR[2] = {}, aU[2] = {}, aIN[2] = {}, aHN[2] = {};
      const int colA = (wave * 4 + p * 2) * 16;
      #pragma unroll
      for (int k0 = 0; k0 < DZD; k0 += 32) {            // gi = z @ Wih^T
        v16bf a = fragA(sh_z, nl, LDZ, k0, half);
        #pragma unroll
        for (int q = 0; q < 2; ++q) {
          int col = colA + q * 16 + nl;
          aR[q]  = wmma_bf(a, fragB(wih,           col, DZD, k0, half), aR[q]);
          aU[q]  = wmma_bf(a, fragB(wih,   DHD   + col, DZD, k0, half), aU[q]);
          aIN[q] = wmma_bf(a, fragB(wih, 2*DHD   + col, DZD, k0, half), aIN[q]);
        }
      }
      #pragma unroll
      for (int k0 = 0; k0 < DHD; k0 += 32) {            // gh = h @ Whh^T
        v16bf a = fragA(hb, nl, LDH, k0, half);
        #pragma unroll
        for (int q = 0; q < 2; ++q) {
          int col = colA + q * 16 + nl;
          aR[q]  = wmma_bf(a, fragB(whh,           col, DHD, k0, half), aR[q]);
          aU[q]  = wmma_bf(a, fragB(whh,   DHD   + col, DHD, k0, half), aU[q]);
          aHN[q] = wmma_bf(a, fragB(whh, 2*DHD   + col, DHD, k0, half), aHN[q]);
        }
      }
      #pragma unroll
      for (int q = 0; q < 2; ++q) {
        int col0 = colA + q * 16;
        float brr = bih[col0 + nl]           + bhh[col0 + nl];
        float buu = bih[DHD + col0 + nl]     + bhh[DHD + col0 + nl];
        float bin = bih[2 * DHD + col0 + nl];
        float bhn = bhh[2 * DHD + col0 + nl];
        #pragma unroll
        for (int i = 0; i < 8; ++i) {
          int m = i + 8 * half;
          float r  = sigmoidf_(aR[q][i] + brr);
          float u  = sigmoidf_(aU[q][i] + buu);
          float ng = tanhf((aIN[q][i] + bin) + r * (aHN[q][i] + bhn));
          float ho = (float)hb[m * LDH + col0 + nl];
          __bf16 hv = (__bf16)((1.0f - u) * ng + u * ho);
          hn[m * LDH + col0 + nl] = hv;
          hseq[((size_t)(b0 + m) * T_SZ + t) * DHD + col0 + nl] = hv;
        }
      }
    }
    __syncthreads();
    cur ^= 1;
  }
}

// ---------------------------------------------------------------------------
// Kernel 2: decoupled emitter — batched GEMM over all B*T rows.
// grid = B*T/16 = 8192 WGs; dx weights (0.79 MB bf16) stay hot in L2/WGP$.
// ---------------------------------------------------------------------------
__global__ __launch_bounds__(256, 1) void emitter_kernel(
    const __bf16* __restrict__ hseq,                                  // [B*T, DH]
    const __bf16* __restrict__ wx1, const float* __restrict__ bx1,    // [DDX,DH]
    const __bf16* __restrict__ wx2, const float* __restrict__ bx2,    // [DX,DDX]
    float* __restrict__ out)                                          // [B*T, DX]
{
  __shared__ __bf16 sh_hid2[16 * LDZ];

  const int tid  = threadIdx.x;
  const int lane = tid & 31;
  const int wave = tid >> 5;
  const int half = lane >> 4;
  const int nl   = lane & 15;
  const size_t row0 = (size_t)blockIdx.x * 16;
  const __bf16* hrows = hseq + row0 * DHD;

  // hid2 = relu(h @ dx_W1^T + b1)   [16 x DDX], K=DH; A straight from global,
  // loaded once per k-step and shared by both column tiles.
  {
    v8f acc[2] = {};
    #pragma unroll
    for (int k0 = 0; k0 < DHD; k0 += 32) {
      v16bf a = fragA(hrows, nl, DHD, k0, half);
      #pragma unroll
      for (int jt = 0; jt < 2; ++jt)
        acc[jt] = wmma_bf(a, fragB(wx1, (wave * 2 + jt) * 16 + nl, DHD, k0, half), acc[jt]);
    }
    #pragma unroll
    for (int jt = 0; jt < 2; ++jt) {
      int col0 = (wave * 2 + jt) * 16;
      float bias = bx1[col0 + nl];
      #pragma unroll
      for (int i = 0; i < 8; ++i) {
        int m = i + 8 * half;
        float v = acc[jt][i] + bias;
        sh_hid2[m * LDZ + col0 + nl] = (__bf16)(v > 0.0f ? v : 0.0f);
      }
    }
  }
  __syncthreads();

  // x_ps = sigmoid(hid2 @ dx_W2^T + b2)   [16 x DX], K=DDX.
  // One A-load per k-step feeds all 8 of the wave's column tiles.
  {
    v8f acc[8] = {};
    #pragma unroll
    for (int k0 = 0; k0 < DDX; k0 += 32) {
      v16bf a = fragA(sh_hid2, nl, LDZ, k0, half);
      #pragma unroll
      for (int jt = 0; jt < 8; ++jt)
        acc[jt] = wmma_bf(a, fragB(wx2, (wave * 8 + jt) * 16 + nl, DDX, k0, half), acc[jt]);
    }
    #pragma unroll
    for (int jt = 0; jt < 8; ++jt) {
      int col0 = (wave * 8 + jt) * 16;
      float bias = bx2[col0 + nl];
      #pragma unroll
      for (int i = 0; i < 8; ++i) {
        int m = i + 8 * half;
        out[(row0 + m) * DXO + col0 + nl] = sigmoidf_(acc[jt][i] + bias);
      }
    }
  }
}

// ---------------------------------------------------------------------------
// Launch
// ---------------------------------------------------------------------------
extern "C" void kernel_launch(void* const* d_in, const int* in_sizes, int n_in,
                              void* d_out, int out_size, void* d_ws, size_t ws_size,
                              hipStream_t stream) {
  const float* eps  = (const float*)d_in[0];
  const float* W1   = (const float*)d_in[1];
  const float* b1   = (const float*)d_in[2];
  const float* Wloc = (const float*)d_in[3];
  const float* bloc = (const float*)d_in[4];
  const float* Wscl = (const float*)d_in[5];
  const float* bscl = (const float*)d_in[6];
  const float* Wih  = (const float*)d_in[7];
  const float* Whh  = (const float*)d_in[8];
  const float* bih  = (const float*)d_in[9];
  const float* bhh  = (const float*)d_in[10];
  const float* Wx1  = (const float*)d_in[11];
  const float* bx1  = (const float*)d_in[12];
  const float* Wx2  = (const float*)d_in[13];
  const float* bx2  = (const float*)d_in[14];
  const float* h0   = (const float*)d_in[15];
  float* out = (float*)d_out;

  // Workspace layout (bf16): 7 weight arrays (3.67 MB) then hseq (134 MB).
  __bf16* ws = (__bf16*)d_ws;
  const size_t nW1   = (size_t)DDZ * DHD;
  const size_t nWloc = (size_t)DZD * DDZ;
  const size_t nWscl = (size_t)DZD * DDZ;
  const size_t nWih  = (size_t)3 * DHD * DZD;
  const size_t nWhh  = (size_t)3 * DHD * DHD;
  const size_t nWx1  = (size_t)DDX * DHD;
  const size_t nWx2  = (size_t)DXO * DDX;
  __bf16* wz1  = ws;               size_t off = nW1;
  __bf16* wloc = ws + off;         off += nWloc;
  __bf16* wscl = ws + off;         off += nWscl;
  __bf16* wih  = ws + off;         off += nWih;
  __bf16* whh  = ws + off;         off += nWhh;
  __bf16* wx1  = ws + off;         off += nWx1;
  __bf16* wx2  = ws + off;         off += nWx2;
  __bf16* hseq = ws + off;         // [B,T,DH] bf16

  auto cvt = [&](const float* s, __bf16* d, size_t n) {
    int blocks = (int)((n + 255) / 256);
    cvt_bf16_kernel<<<blocks, 256, 0, stream>>>(s, d, (int)n);
  };
  cvt(W1, wz1, nW1);
  cvt(Wloc, wloc, nWloc);
  cvt(Wscl, wscl, nWscl);
  cvt(Wih, wih, nWih);
  cvt(Whh, whh, nWhh);
  cvt(Wx1, wx1, nWx1);
  cvt(Wx2, wx2, nWx2);

  // Serial scan: 32 persistent workgroups.
  dmm_scan_kernel<<<B_SZ / 16, 256, 0, stream>>>(
      eps, wz1, b1, wloc, bloc, wscl, bscl,
      wih, whh, bih, bhh, h0, hseq);

  // Decoupled emitter: 8192 workgroups, chip-wide parallel.
  emitter_kernel<<<(B_SZ * T_SZ) / 16, 256, 0, stream>>>(
      hseq, wx1, bx1, wx2, bx2, out);
}
